// LSTM_78005196030512
// MI455X (gfx1250) — matrix operand order repaired and checked by reading, not compiled, not run on hardware
//
#include <hip/hip_runtime.h>
#include <math.h>

// ---------------------------------------------------------------------------
// 2-layer LSTM LM forward for MI455X (gfx1250, wave32, WMMA bf16 -> f32)
//  - big input-side GEMMs: software-pipelined LDS staging + v_wmma bf16
//  - recurrent steps: TDM (tensor_load_to_lds) double-buffered tile staging,
//    s_wait_tensorcnt, fused LSTM cell update
// ---------------------------------------------------------------------------

#define DIM   1024
#define GATES 4096
#define BATCH 64
#define SEQ   256
#define EMB   256
#define VOCAB 256
#define MROWS (BATCH * SEQ)   // 16384

typedef __attribute__((ext_vector_type(16))) __bf16 v16bf;
typedef __attribute__((ext_vector_type(8)))  __bf16 v8bf;
typedef __attribute__((ext_vector_type(8)))  float  v8f;
typedef __attribute__((ext_vector_type(4)))  unsigned int u32x4;
typedef __attribute__((ext_vector_type(8)))  int i32x8;
typedef __attribute__((ext_vector_type(4)))  int i32x4;

__device__ __forceinline__ v8f wmma_bf16(v16bf a, v16bf b, v8f c) {
  return __builtin_amdgcn_wmma_f32_16x16x32_bf16(
      /*neg_a=*/false, a, /*neg_b=*/false, b,
      /*c_mod=*/(short)0, c, /*reuse_a=*/false, /*reuse_b=*/false);
}

// A-fragment (16x32 bf16 tile in LDS, row stride `ld` halves).
// ISA 7.12.2: lanes 0-15: M=lane, K={0..7,16..23}; lanes 16-31: K={8..15,24..31}
__device__ __forceinline__ v16bf frag_a_lds(const __bf16* base, int ld) {
  const int lane = threadIdx.x & 31;
  const int m  = lane & 15;
  const int hi = lane >> 4;
  const v8bf* p0 = (const v8bf*)(base + m * ld + hi * 8);
  const v8bf* p1 = (const v8bf*)(base + m * ld + 16 + hi * 8);
  v8bf lo = *p0, hh = *p1;
  return __builtin_shufflevector(lo, hh, 0,1,2,3,4,5,6,7,8,9,10,11,12,13,14,15);
}

// B-fragment (32x16): LDS stores W row-major [n][k] (16 rows x 32 k).
// lanes 0-15: n=lane K={0..7,16..23}; lanes 16-31: n=lane-16 K={8..15,24..31}
__device__ __forceinline__ v16bf frag_b_lds(const __bf16* base, int ld) {
  const int lane = threadIdx.x & 31;
  const int n  = lane & 15;
  const int hi = lane >> 4;
  const v8bf* p0 = (const v8bf*)(base + n * ld + hi * 8);
  const v8bf* p1 = (const v8bf*)(base + n * ld + 16 + hi * 8);
  v8bf lo = *p0, hh = *p1;
  return __builtin_shufflevector(lo, hh, 0,1,2,3,4,5,6,7,8,9,10,11,12,13,14,15);
}

__device__ __forceinline__ float sigmoidf_(float x) {
  return 1.0f / (1.0f + __expf(-x));
}

// ---------------------------------------------------------------------------
// Tensor Data Mover: 2-D tile load (rows x 32 bf16 elems) global -> LDS.
// D# packing per CDNA5 ISA 8.3/8.4: group0 = {flags/count, lds_addr,
// global_addr lo, global_addr hi | type=2}; group1 carries data_size=2B,
// tensor dims (generous in-bounds), tile dims, dim0 stride. Groups 2/3 unused.
// lds_byte is a byte offset in the static group segment (single __shared__
// allocation -> segment-relative offsets start at 0).
// ---------------------------------------------------------------------------
__device__ __forceinline__ void tdm_load_tile_2d(unsigned lds_byte,
                                                 const __bf16* g,
                                                 int rows, int stride_elems) {
  unsigned long long ga = (unsigned long long)g;
  u32x4 g0;
  g0.x = 1u;                                                // count=1 (valid)
  g0.y = lds_byte;                                          // lds_addr
  g0.z = (unsigned)ga;                                      // global_addr lo
  g0.w = (unsigned)((ga >> 32) & 0x01ffffffull) | (2u << 30); // hi + type=2
  const unsigned td0 = 1u << 20, td1 = 1u << 20;            // tensor dims
  i32x8 g1;
  g1[0] = (int)(1u << 16);                                  // data_size = 2B
  g1[1] = (int)((td0 & 0xffffu) << 16);                     // tensor_dim0 lo16
  g1[2] = (int)(((td0 >> 16) & 0xffffu) | ((td1 & 0xffffu) << 16));
  g1[3] = (int)(((td1 >> 16) & 0xffffu) | (32u << 16));     // tile_dim0 = 32
  g1[4] = rows & 0xffff;                                    // tile_dim1, dim2=0
  g1[5] = stride_elems;                                     // dim0_stride lo32
  g1[6] = 0;
  g1[7] = 0;
  i32x4 z4 = {0, 0, 0, 0};
#if defined(__clang_major__) && __clang_major__ >= 23
  i32x8 z8 = {0, 0, 0, 0, 0, 0, 0, 0};
  __builtin_amdgcn_tensor_load_to_lds(g0, g1, z4, z4, z8, 0);
#else
  __builtin_amdgcn_tensor_load_to_lds(g0, g1, z4, z4, 0);
#endif
}

// ---------------------------------------------------------------------------
// weight_norm (dim=0): w[r,:] = g[r] * v[r,:] / ||v[r,:]||, store bf16
// ---------------------------------------------------------------------------
__global__ void k_weightnorm_bf16(const float* __restrict__ v,
                                  const float* __restrict__ g,
                                  __bf16* __restrict__ w, int rowlen) {
  __shared__ float red[256];
  const int row = blockIdx.x;
  const float* vr = v + (size_t)row * rowlen;
  float s = 0.f;
  for (int j = threadIdx.x; j < rowlen; j += 256) { float x = vr[j]; s += x * x; }
  red[threadIdx.x] = s;
  __syncthreads();
  for (int off = 128; off > 0; off >>= 1) {
    if ((int)threadIdx.x < off) red[threadIdx.x] += red[threadIdx.x + off];
    __syncthreads();
  }
  const float scale = g[row] / sqrtf(red[0]);
  __bf16* wr = w + (size_t)row * rowlen;
  for (int j = threadIdx.x; j < rowlen; j += 256)
    wr[j] = (__bf16)(vr[j] * scale);
}

__global__ void k_f32_to_bf16(const float* __restrict__ in,
                              __bf16* __restrict__ out, int n) {
  int i = blockIdx.x * 256 + threadIdx.x;
  if (i < n) out[i] = (__bf16)in[i];
}

__global__ void k_bias_sum(const float* __restrict__ a,
                           const float* __restrict__ b,
                           float* __restrict__ out, int n) {
  int i = blockIdx.x * 256 + threadIdx.x;
  if (i < n) out[i] = a[i] + b[i];
}

// embedding gather: e[row, j] = emb[x[row], j], row = b*T + t
__global__ void k_embed(const int* __restrict__ x, const float* __restrict__ emb,
                        __bf16* __restrict__ e) {
  int i = blockIdx.x * 256 + threadIdx.x;           // < MROWS*EMB
  int row = i >> 8;
  int col = i & 255;
  e[i] = (__bf16)emb[(size_t)x[row] * EMB + col];
}

// ---------------------------------------------------------------------------
// Generic bf16 WMMA GEMM: C[M,N] = A[M,K] * Bm[N,K]^T + bias[N]
// Block tile 128(M) x 64(N), 256 threads (8 wave32).
// Software pipeline: global loads for K-tile k+1 issued before computing k.
// ---------------------------------------------------------------------------
template <typename OutT>
__global__ void k_gemm_bf16(const __bf16* __restrict__ A,
                            const __bf16* __restrict__ Bm,
                            const float* __restrict__ bias,
                            OutT* __restrict__ C, int M, int N, int K) {
  __shared__ __bf16 sA[128 * 32];
  __shared__ __bf16 sB[64 * 32];

  const int tid  = threadIdx.x;
  const int w    = tid >> 5;
  const int lane = tid & 31;
  const int gm0  = blockIdx.y * 128;
  const int gn0  = blockIdx.x * 64;

  const int ra = tid >> 1, ca = (tid & 1) * 16;   // A: 16 halves / thread
  const int rb = tid >> 2, cb = (tid & 3) * 8;    // B: 8 halves / thread

  v8f acc[4] = {};
  v8bf rA0, rA1, rB;

  // prologue: fetch tile 0
  {
    const v8bf* srcA = (const v8bf*)(A + (size_t)(gm0 + ra) * K + ca);
    rA0 = srcA[0]; rA1 = srcA[1];
    rB = *(const v8bf*)(Bm + (size_t)(gn0 + rb) * K + cb);
  }
  ((v8bf*)(sA + ra * 32 + ca))[0] = rA0;
  ((v8bf*)(sA + ra * 32 + ca))[1] = rA1;
  *(v8bf*)(sB + rb * 32 + cb) = rB;
  __syncthreads();

  for (int ko = 0; ko < K; ko += 32) {
    const bool more = (ko + 32) < K;
    if (more) {  // issue next tile's global loads under the WMMA clause
      const v8bf* srcA = (const v8bf*)(A + (size_t)(gm0 + ra) * K + ko + 32 + ca);
      rA0 = srcA[0]; rA1 = srcA[1];
      rB = *(const v8bf*)(Bm + (size_t)(gn0 + rb) * K + ko + 32 + cb);
    }

    v16bf af = frag_a_lds(sA + w * 16 * 32, 32);
#pragma unroll
    for (int nt = 0; nt < 4; ++nt) {
      v16bf bf = frag_b_lds(sB + nt * 16 * 32, 32);
      acc[nt] = wmma_bf16(af, bf, acc[nt]);
    }
    __syncthreads();

    if (more) {
      ((v8bf*)(sA + ra * 32 + ca))[0] = rA0;
      ((v8bf*)(sA + ra * 32 + ca))[1] = rA1;
      *(v8bf*)(sB + rb * 32 + cb) = rB;
    }
    __syncthreads();
  }

#pragma unroll
  for (int nt = 0; nt < 4; ++nt) {
    int col = gn0 + nt * 16 + (lane & 15);
    float bv = bias[col];
#pragma unroll
    for (int r = 0; r < 8; ++r) {
      int row = gm0 + w * 16 + ((lane & 16) ? r + 8 : r);
      C[(size_t)row * N + col] = (OutT)(acc[nt][r] + bv);
    }
  }
}

// ---------------------------------------------------------------------------
// Fused LSTM step, one launch per timestep, 16 blocks x 256 threads.
// Block bk owns gate slice d in [bk*64, bk*64+64) across all 4 gates.
//   gates[64B x 256N] = h_in[64 x 1024] @ Whh[256 rows]^T  (WMMA)
// Tile staging done by the Tensor Data Mover, double-buffered:
//   LDS bytes: A ping/pong at 0 / 4K, B ping/pong at 8K / 24K;
//   the whole 64KB region is reused as the f32 gate exchange buffer after.
// NOTE: buffer pointers are computed arithmetically from `cur` (no pointer
// arrays) to avoid static initializers containing LDS addrspacecasts.
// ---------------------------------------------------------------------------
__global__ void k_lstm_step(const __bf16* __restrict__ xg,   // [MROWS][GATES]
                            const __bf16* __restrict__ Whh,  // [GATES][DIM]
                            const __bf16* __restrict__ h_in, // [B][DIM]
                            __bf16* __restrict__ h_out,      // [B][DIM]
                            float* __restrict__ cbuf,        // [B][DIM]
                            float* __restrict__ h_f32,       // [B][DIM]
                            __bf16* __restrict__ hs,         // [MROWS][DIM]
                            int t) {
  __shared__ float smem[64 * 256];                 // 64 KB
  __bf16* const sbase = (__bf16*)smem;
  // half-element offsets: A buf = cur*2048 ; B buf = 4096 + cur*8192
  // LDS byte offsets:     A buf = cur*4096 ; B buf = 8192 + cur*16384

  const int tid  = threadIdx.x;
  const int w    = tid >> 5;
  const int lane = tid & 31;
  const int s    = blockIdx.x * 64;                // d-slice base

  // warm this step's xg gate slice toward L2 while the GEMM runs
  {
    size_t xrow = ((size_t)(tid >> 2) * SEQ + t) * GATES;
#pragma unroll
    for (int q = 0; q < 4; ++q)
      __builtin_prefetch(xg + xrow + q * DIM + s + (tid & 3) * 16, 0, 1);
  }

  // prologue: TDM fills buffer 0 (A tile + 4 gate-row tiles)
  if (w == 0) {
    tdm_load_tile_2d(0u, h_in, 64, DIM);
#pragma unroll
    for (int q = 0; q < 4; ++q)
      tdm_load_tile_2d(8192u + q * 4096u,
                       Whh + ((size_t)(q * DIM + s)) * DIM, 64, DIM);
    __builtin_amdgcn_s_wait_tensorcnt(0);
  }
  __syncthreads();

  v8f acc[8] = {};                                 // wave w: nt=2w,2w+1; mt 0..3
  int cur = 0;
  for (int ko = 0; ko < DIM; ko += 32, cur ^= 1) {
    const int  nxt  = cur ^ 1;
    const bool more = (ko + 32) < DIM;
    if (more && w == 0) {                          // DMA next tiles during WMMA
      tdm_load_tile_2d((unsigned)(nxt * 4096), h_in + ko + 32, 64, DIM);
#pragma unroll
      for (int q = 0; q < 4; ++q)
        tdm_load_tile_2d((unsigned)(8192 + nxt * 16384 + q * 4096),
                         Whh + ((size_t)(q * DIM + s)) * DIM + ko + 32, 64, DIM);
    }

    const __bf16* sAcur = sbase + cur * 2048;
    const __bf16* sBcur = sbase + 4096 + cur * 8192;
    v16bf af[4];
#pragma unroll
    for (int mt = 0; mt < 4; ++mt)
      af[mt] = frag_a_lds(sAcur + mt * 16 * 32, 32);
    v16bf bf0 = frag_b_lds(sBcur + (2 * w) * 16 * 32, 32);
    v16bf bf1 = frag_b_lds(sBcur + (2 * w + 1) * 16 * 32, 32);
#pragma unroll
    for (int mt = 0; mt < 4; ++mt) acc[mt]     = wmma_bf16(af[mt], bf0, acc[mt]);
#pragma unroll
    for (int mt = 0; mt < 4; ++mt) acc[4 + mt] = wmma_bf16(af[mt], bf1, acc[4 + mt]);

    if (more && w == 0) __builtin_amdgcn_s_wait_tensorcnt(0);
    __syncthreads();
  }

  // spill accumulators to LDS gates[64 rows(b)][256 cols(n_local)]
#pragma unroll
  for (int a = 0; a < 8; ++a) {
    int nt = 2 * w + (a >> 2);
    int mt = a & 3;
    int col = nt * 16 + (lane & 15);
#pragma unroll
    for (int r = 0; r < 8; ++r) {
      int row = mt * 16 + ((lane & 16) ? r + 8 : r);
      smem[row * 256 + col] = acc[a][r];
    }
  }
  __syncthreads();

  // fused cell update: 4096 (b,d) pairs, 16 per thread
  for (int idx = tid; idx < 4096; idx += 256) {
    int b  = idx >> 6;
    int dd = idx & 63;
    int d  = s + dd;
    size_t xbase = ((size_t)b * SEQ + t) * GATES;
    float ig = smem[b * 256 +   0 + dd] + (float)xg[xbase + 0 * DIM + s + dd];
    float fg = smem[b * 256 +  64 + dd] + (float)xg[xbase + 1 * DIM + s + dd];
    float gg = smem[b * 256 + 128 + dd] + (float)xg[xbase + 2 * DIM + s + dd];
    float og = smem[b * 256 + 192 + dd] + (float)xg[xbase + 3 * DIM + s + dd];

    size_t ci = (size_t)b * DIM + d;
    float c_new = sigmoidf_(fg) * cbuf[ci] + sigmoidf_(ig) * tanhf(gg);
    float h_new = sigmoidf_(og) * tanhf(c_new);

    cbuf[ci]  = c_new;
    h_f32[ci] = h_new;
    h_out[ci] = (__bf16)h_new;
    hs[((size_t)b * SEQ + t) * DIM + d] = (__bf16)h_new;
  }
}

// state init / finalize ------------------------------------------------------
__global__ void k_state_init(const float* __restrict__ h0,
                             const float* __restrict__ c0,
                             __bf16* __restrict__ hbf,
                             float* __restrict__ cbuf,
                             float* __restrict__ hf) {
  int i = blockIdx.x * 256 + threadIdx.x;      // < B*DIM
  float h = h0[i];
  hbf[i] = (__bf16)h;
  hf[i]  = h;
  cbuf[i] = c0[i];
}

__global__ void k_finalize(const float* __restrict__ hf,
                           const float* __restrict__ cbuf,
                           float* __restrict__ out_h,
                           float* __restrict__ out_c) {
  int i = blockIdx.x * 256 + threadIdx.x;      // < B*DIM
  out_h[i] = hf[i];
  out_c[i] = cbuf[i];
}

// ---------------------------------------------------------------------------
extern "C" void kernel_launch(void* const* d_in, const int* in_sizes, int n_in,
                              void* d_out, int out_size, void* d_ws, size_t ws_size,
                              hipStream_t stream) {
  (void)in_sizes; (void)n_in; (void)out_size; (void)ws_size;

  const int*   x      = (const int*)  d_in[0];
  const float* h0     = (const float*)d_in[1];   // [2,B,DIM]
  const float* c0     = (const float*)d_in[2];
  const float* emb    = (const float*)d_in[3];
  const float* v_ih0  = (const float*)d_in[4];
  const float* g_ih0  = (const float*)d_in[5];
  const float* v_hh0  = (const float*)d_in[6];
  const float* g_hh0  = (const float*)d_in[7];
  const float* b_ih0  = (const float*)d_in[8];
  const float* b_hh0  = (const float*)d_in[9];
  const float* W_ih1  = (const float*)d_in[10];
  const float* W_hh1  = (const float*)d_in[11];
  const float* b_ih1  = (const float*)d_in[12];
  const float* b_hh1  = (const float*)d_in[13];
  const float* W_out  = (const float*)d_in[14];
  const float* b_out  = (const float*)d_in[15];

  // workspace carve-up (256B aligned)
  size_t off = 0;
  auto alloc = [&](size_t bytes) -> void* {
    void* p = (char*)d_ws + off;
    off += (bytes + 255) & ~(size_t)255;
    return p;
  };
  __bf16* Wih0 = (__bf16*)alloc((size_t)GATES * EMB * 2);
  __bf16* Whh0 = (__bf16*)alloc((size_t)GATES * DIM * 2);
  __bf16* Wih1 = (__bf16*)alloc((size_t)GATES * DIM * 2);
  __bf16* Whh1 = (__bf16*)alloc((size_t)GATES * DIM * 2);
  __bf16* Wout = (__bf16*)alloc((size_t)VOCAB * DIM * 2);
  float*  b0   = (float*) alloc((size_t)GATES * 4);
  float*  b1   = (float*) alloc((size_t)GATES * 4);
  __bf16* ebf  = (__bf16*)alloc((size_t)MROWS * EMB * 2);
  __bf16* xg   = (__bf16*)alloc((size_t)MROWS * GATES * 2);   // 128 MB
  __bf16* h1s  = (__bf16*)alloc((size_t)MROWS * DIM * 2);
  __bf16* h2s  = (__bf16*)alloc((size_t)MROWS * DIM * 2);
  __bf16* hp0  = (__bf16*)alloc((size_t)BATCH * DIM * 2);     // ping
  __bf16* hp1  = (__bf16*)alloc((size_t)BATCH * DIM * 2);     // pong
  float*  cbuf = (float*) alloc((size_t)BATCH * DIM * 4);
  float*  hf32 = (float*) alloc((size_t)BATCH * DIM * 4);

  float* out_logits = (float*)d_out;                          // [B,T,VOCAB]
  float* out_h      = out_logits + (size_t)MROWS * VOCAB;     // [2,B,DIM]
  float* out_c      = out_h + 2 * (size_t)BATCH * DIM;

  __bf16* hbuf[2] = {hp0, hp1};

  // ---- weight prep ----
  k_weightnorm_bf16<<<GATES, 256, 0, stream>>>(v_ih0, g_ih0, Wih0, EMB);
  k_weightnorm_bf16<<<GATES, 256, 0, stream>>>(v_hh0, g_hh0, Whh0, DIM);
  k_f32_to_bf16<<<(GATES * DIM + 255) / 256, 256, 0, stream>>>(W_ih1, Wih1, GATES * DIM);
  k_f32_to_bf16<<<(GATES * DIM + 255) / 256, 256, 0, stream>>>(W_hh1, Whh1, GATES * DIM);
  k_f32_to_bf16<<<(VOCAB * DIM + 255) / 256, 256, 0, stream>>>(W_out, Wout, VOCAB * DIM);
  k_bias_sum<<<GATES / 256, 256, 0, stream>>>(b_ih0, b_hh0, b0, GATES);
  k_bias_sum<<<GATES / 256, 256, 0, stream>>>(b_ih1, b_hh1, b1, GATES);

  // ---- embedding ----
  k_embed<<<(MROWS * EMB) / 256, 256, 0, stream>>>(x, emb, ebf);

  // ================= layer 0 =================
  {
    dim3 grid(GATES / 64, MROWS / 128);
    k_gemm_bf16<__bf16><<<grid, 256, 0, stream>>>(ebf, Wih0, b0, xg,
                                                  MROWS, GATES, EMB);
  }
  k_state_init<<<(BATCH * DIM) / 256, 256, 0, stream>>>(h0, c0, hbuf[0], cbuf, hf32);
  for (int t = 0; t < SEQ; ++t) {
    k_lstm_step<<<16, 256, 0, stream>>>(xg, Whh0, hbuf[t & 1], hbuf[(t + 1) & 1],
                                        cbuf, hf32, h1s, t);
  }
  k_finalize<<<(BATCH * DIM) / 256, 256, 0, stream>>>(hf32, cbuf, out_h, out_c);

  // ================= layer 1 =================
  {
    dim3 grid(GATES / 64, MROWS / 128);
    k_gemm_bf16<__bf16><<<grid, 256, 0, stream>>>(h1s, Wih1, b1, xg,
                                                  MROWS, GATES, DIM);
  }
  k_state_init<<<(BATCH * DIM) / 256, 256, 0, stream>>>(
      h0 + (size_t)BATCH * DIM, c0 + (size_t)BATCH * DIM, hbuf[0], cbuf, hf32);
  for (int t = 0; t < SEQ; ++t) {
    k_lstm_step<<<16, 256, 0, stream>>>(xg, Whh1, hbuf[t & 1], hbuf[(t + 1) & 1],
                                        cbuf, hf32, h2s, t);
  }
  k_finalize<<<(BATCH * DIM) / 256, 256, 0, stream>>>(
      hf32, cbuf, out_h + (size_t)BATCH * DIM, out_c + (size_t)BATCH * DIM);

  // ---- readout: logits = h2s @ W_out^T + b_out ----
  {
    dim3 grid(VOCAB / 64, MROWS / 128);
    k_gemm_bf16<float><<<grid, 256, 0, stream>>>(h2s, Wout, b_out, out_logits,
                                                 MROWS, VOCAB, DIM);
  }
}
